// MultiHeadStructureAttention_59992103190952
// MI455X (gfx1250) — compile-verified, matrix operand order checked by reading, not run on hardware
//
#include <hip/hip_runtime.h>
#include <hip/hip_bf16.h>

typedef __attribute__((ext_vector_type(16))) _Float16     v16h;
typedef __attribute__((ext_vector_type(8)))  float        v8f;
typedef __attribute__((ext_vector_type(4)))  unsigned int v4u;
typedef __attribute__((ext_vector_type(8)))  int          v8i;
typedef __attribute__((ext_vector_type(4)))  int          v4i;

#define S_LEN   2048
#define D_MODEL 512
#define NHEAD   8
#define HDIM    64
#define BATCH   2

#if defined(__has_builtin)
#if __has_builtin(__builtin_amdgcn_tensor_load_to_lds) && __has_builtin(__builtin_amdgcn_s_wait_tensorcnt)
#define HAVE_TDM 1
#endif
#endif
#ifndef HAVE_TDM
#define HAVE_TDM 0
#endif

// Build a 16x32 f16 A-operand (or K-contiguous B-operand) fragment from a
// 32-element contiguous K-run. Wave32 layout per CDNA5 ISA 7.12.2:
//   lanes 0-15 : elems 0-7 -> K0-7,  elems 8-15 -> K16-23
//   lanes 16-31: elems 0-7 -> K8-15, elems 8-15 -> K24-31
__device__ __forceinline__ v16h load_frag_row(const _Float16* row, int lane) {
    const int kb = (lane & 16) ? 8 : 0;
    v16h r;
    uint4* rp = reinterpret_cast<uint4*>(&r);
    rp[0] = *reinterpret_cast<const uint4*>(row + kb);
    rp[1] = *reinterpret_cast<const uint4*>(row + kb + 16);
    return r;
}

#if HAVE_TDM
// LDS byte offset of a __shared__ object (generic -> addrspace(3) -> int)
__device__ __forceinline__ unsigned lds_off(const void* p) {
    return (unsigned)(unsigned long long)
        (__attribute__((address_space(3))) const void*)p;
}

// TDM: load a 32(key) x 64(dk) f16 tile, row stride 64 elems in memory,
// into LDS rows padded to 72 f16 (pad 4 DWORDs after every 32 DWORDs).
__device__ __forceinline__ void tdm_load_k32(const _Float16* gsrc, unsigned dst) {
    unsigned long long ga = (unsigned long long)(const void*)gsrc;
    v4u g0;
    g0[0] = 1u;                                        // count=1, user mode
    g0[1] = dst;                                       // lds_addr
    g0[2] = (unsigned)(ga & 0xffffffffu);              // global_addr lo
    g0[3] = (unsigned)((ga >> 32) & 0x01ffffffu) | (2u << 30);  // addr hi | type=2
    v8i g1;
    g1[0] = (int)((1u << 16) |                         // data_size = 2 bytes
                  (1u << 20) |                         // pad_enable
                  (4u << 22) |                         // pad_interval: 32 DWORDs
                  (3u << 25));                         // pad_amount: 4 DWORDs
    g1[1] = (int)(64u << 16);                          // tensor_dim0 = 64
    g1[2] = (int)(32u << 16);                          // tensor_dim1 = 32
    g1[3] = (int)(64u << 16);                          // tile_dim0 = 64
    g1[4] = 32;                                        // tile_dim1 = 32, tile_dim2 = 0
    g1[5] = 64;                                        // tensor_dim0_stride = 64
    g1[6] = 0;
    g1[7] = 0;
    v4i z4 = {0, 0, 0, 0};
    v8i z8 = {0, 0, 0, 0, 0, 0, 0, 0};
    // clang-23 / therock-10.0 6-arg form:
    // (g0, g1, g2, g3, <extra group>, cache-policy)
    __builtin_amdgcn_tensor_load_to_lds(g0, g1, z4, z4, z8, 0);
}
#endif

// ---------------------------------------------------------------------------
// Projection GEMM: [4096 x 512] @ [512 x 512] (+bias, *oscale)
// modes 0/1/2: A fp32 in, write f16 out in [b][h][s][d] layout (Q/K/V)
// mode 3     : A f16 in (ctx), write fp32 out row-major (final output)
// Block: 256 thr = 8 waves arranged 4(M) x 2(N); block tile 64M x 64N;
// each wave: 16x32 slab = 2 WMMAs sharing one A fragment per K-step.
// ---------------------------------------------------------------------------
__global__ void __launch_bounds__(256)
proj_kernel(const float* __restrict__ A32, const _Float16* __restrict__ A16,
            const float* __restrict__ W, const float* __restrict__ bias,
            _Float16* __restrict__ out16, float* __restrict__ out32,
            float oscale, int mode)
{
    __shared__ __align__(16) _Float16 ldsA[64][40];   // [m][k]
    __shared__ __align__(16) _Float16 ldsBT[64][40];  // [n][k] (W transposed)

    const int tid = threadIdx.x, lane = tid & 31, wave = tid >> 5;
    const int wm = wave & 3, wn = wave >> 2;
    const int m0 = blockIdx.x * 64;
    const int n0 = blockIdx.y * 64;
    const int col = lane & 15;

    v8f acc0 = {}, acc1 = {};

    for (int kb = 0; kb < D_MODEL; kb += 32) {
        __syncthreads();
        {   // stage A 64x32 (fp32 -> f16 unless mode 3)
            const int m = tid >> 2, kc = (tid & 3) * 8;
            if (mode == 3) {
                *reinterpret_cast<uint4*>(&ldsA[m][kc]) =
                    *reinterpret_cast<const uint4*>(&A16[(size_t)(m0 + m) * D_MODEL + kb + kc]);
            } else {
                const float* src = &A32[(size_t)(m0 + m) * D_MODEL + kb + kc];
                float4 f0 = *reinterpret_cast<const float4*>(src);
                float4 f1 = *reinterpret_cast<const float4*>(src + 4);
                __align__(16) _Float16 t[8];
                t[0] = (_Float16)f0.x; t[1] = (_Float16)f0.y;
                t[2] = (_Float16)f0.z; t[3] = (_Float16)f0.w;
                t[4] = (_Float16)f1.x; t[5] = (_Float16)f1.y;
                t[6] = (_Float16)f1.z; t[7] = (_Float16)f1.w;
                *reinterpret_cast<uint4*>(&ldsA[m][kc]) = *reinterpret_cast<const uint4*>(t);
            }
        }
        {   // stage W 32x64 transposed to [n][k]
            const int kk = tid >> 3, nc = (tid & 7) * 8;
            const float* src = &W[(size_t)(kb + kk) * D_MODEL + n0 + nc];
            float4 f0 = *reinterpret_cast<const float4*>(src);
            float4 f1 = *reinterpret_cast<const float4*>(src + 4);
            ldsBT[nc + 0][kk] = (_Float16)f0.x;
            ldsBT[nc + 1][kk] = (_Float16)f0.y;
            ldsBT[nc + 2][kk] = (_Float16)f0.z;
            ldsBT[nc + 3][kk] = (_Float16)f0.w;
            ldsBT[nc + 4][kk] = (_Float16)f1.x;
            ldsBT[nc + 5][kk] = (_Float16)f1.y;
            ldsBT[nc + 6][kk] = (_Float16)f1.z;
            ldsBT[nc + 7][kk] = (_Float16)f1.w;
        }
        __syncthreads();

        v16h af  = load_frag_row(&ldsA[wm * 16 + col][0], lane);
        v16h bf0 = load_frag_row(&ldsBT[wn * 32 + col][0], lane);
        v16h bf1 = load_frag_row(&ldsBT[wn * 32 + 16 + col][0], lane);
        acc0 = __builtin_amdgcn_wmma_f32_16x16x32_f16(false, af, false, bf0, (short)0, acc0, false, false);
        acc1 = __builtin_amdgcn_wmma_f32_16x16x32_f16(false, af, false, bf1, (short)0, acc1, false, false);
    }

    const int half = (lane & 16) ? 8 : 0;
    const int na = n0 + wn * 32 + col;
    const int nb = na + 16;
    const float ba = bias[na], bb = bias[nb];
#pragma unroll
    for (int v = 0; v < 8; ++v) {
        const int m = m0 + wm * 16 + v + half;
        const float va = (acc0[v] + ba) * oscale;
        const float vb = (acc1[v] + bb) * oscale;
        if (mode == 3) {
            out32[(size_t)m * D_MODEL + na] = va;
            out32[(size_t)m * D_MODEL + nb] = vb;
        } else {
            const int b = m >> 11, s = m & (S_LEN - 1);
            const size_t rowb = (size_t)b * NHEAD * S_LEN * HDIM;
            out16[rowb + (((size_t)(na >> 6) * S_LEN) + s) * HDIM + (na & (HDIM - 1))] = (_Float16)va;
            out16[rowb + (((size_t)(nb >> 6) * S_LEN) + s) * HDIM + (nb & (HDIM - 1))] = (_Float16)vb;
        }
    }
}

// ---------------------------------------------------------------------------
// Fused flash attention with structural bias. 8 waves x 16 q-rows per block.
// Double-buffered 32-key tiles: K staged by the Tensor Data Mover (wave 0),
// V staged transposed by vector loads; one barrier per key-step.
// Scale 1/sqrt(64) is pre-folded into Qf by the projection.
// ---------------------------------------------------------------------------
__global__ void __launch_bounds__(256)
attn_kernel(const _Float16* __restrict__ Qf, const _Float16* __restrict__ Kf,
            const _Float16* __restrict__ Vf, const float* __restrict__ structure,
            _Float16* __restrict__ Cx)
{
    __shared__ __align__(16) _Float16 ldsK[2][32][72];    // [buf][key][dk]
    __shared__ __align__(16) _Float16 ldsVT[2][64][40];   // [buf][dv][key]
    __shared__ __align__(16) _Float16 ldsP[8][16][40];    // per-wave [qrow][key]

    const int tid = threadIdx.x, lane = tid & 31, wave = tid >> 5;
    const int bh = blockIdx.y, b = bh >> 3, h = bh & 7;
    const int q0 = blockIdx.x * 128 + wave * 16;
    const size_t head = (size_t)bh * S_LEN * HDIM;
    const int col  = lane & 15;
    const int half = (lane & 16) ? 8 : 0;
    const int kk = tid >> 3, ch = (tid & 7) * 8;          // staging coords

    // Q fragments (16 queries x 64 dk) kept in registers for the whole pass
    const _Float16* qrow = &Qf[head + (size_t)(q0 + col) * HDIM];
    v16h qa0 = load_frag_row(qrow, lane);
    v16h qa1 = load_frag_row(qrow + 32, lane);

    v8f acc0 = {}, acc1 = {}, acc2 = {}, acc3 = {};
    float mrun[8], lrun[8];
#pragma unroll
    for (int i = 0; i < 8; ++i) { mrun[i] = -1e30f; lrun[i] = 0.0f; }

    // ---- stage tile 0 ----
#if HAVE_TDM
    if (wave == 0) tdm_load_k32(&Kf[head], lds_off(&ldsK[0][0][0]));
#else
    *reinterpret_cast<uint4*>(&ldsK[0][kk][ch]) =
        *reinterpret_cast<const uint4*>(&Kf[head + (size_t)kk * HDIM + ch]);
#endif
    {
        union { uint4 u; _Float16 hlf[8]; } t;
        t.u = *reinterpret_cast<const uint4*>(&Vf[head + (size_t)kk * HDIM + ch]);
#pragma unroll
        for (int i = 0; i < 8; ++i) ldsVT[0][ch + i][kk] = t.hlf[i];
    }
#if HAVE_TDM
    if (wave == 0) __builtin_amdgcn_s_wait_tensorcnt(0);
#endif
    __syncthreads();

    int cur = 0;
    for (int k0 = 0; k0 < S_LEN; k0 += 32) {
        const int nxt = cur ^ 1;
        const bool more = (k0 + 32) < S_LEN;

        // issue next tile's loads before computing on the current one
        union { uint4 u; _Float16 hlf[8]; } vpre;
#if !HAVE_TDM
        uint4 kpre;
#endif
        if (more) {
#if HAVE_TDM
            if (wave == 0)
                tdm_load_k32(&Kf[head + (size_t)(k0 + 32) * HDIM],
                             lds_off(&ldsK[nxt][0][0]));
#else
            kpre = *reinterpret_cast<const uint4*>(&Kf[head + (size_t)(k0 + 32 + kk) * HDIM + ch]);
#endif
            vpre.u = *reinterpret_cast<const uint4*>(&Vf[head + (size_t)(k0 + 32 + kk) * HDIM + ch]);
        }

        // scores: S[16q x 32k] = Qs(16x64) . K^T(64x32)
        v8f s0 = {}, s1 = {};
        {
            v16h bk0 = load_frag_row(&ldsK[cur][col][0], lane);
            v16h bk1 = load_frag_row(&ldsK[cur][col + 16][0], lane);
            s0 = __builtin_amdgcn_wmma_f32_16x16x32_f16(false, qa0, false, bk0, (short)0, s0, false, false);
            s1 = __builtin_amdgcn_wmma_f32_16x16x32_f16(false, qa0, false, bk1, (short)0, s1, false, false);
            bk0 = load_frag_row(&ldsK[cur][col][32], lane);
            bk1 = load_frag_row(&ldsK[cur][col + 16][32], lane);
            s0 = __builtin_amdgcn_wmma_f32_16x16x32_f16(false, qa1, false, bk0, (short)0, s0, false, false);
            s1 = __builtin_amdgcn_wmma_f32_16x16x32_f16(false, qa1, false, bk1, (short)0, s1, false, false);
        }

        // bias + online softmax. C-frag row v lives in one VGPR across a
        // 16-lane half -> xor-shuffle row max; row sum kept as per-lane
        // partials (corr is row-uniform) and reduced once at the end.
#pragma unroll
        for (int v = 0; v < 8; ++v) {
            const int qr = q0 + v + half;
            const float* srow = &structure[((size_t)b * S_LEN + qr) * S_LEN + k0];
            if (more) __builtin_prefetch(srow + 32, 0, 1);  // next tile's bias row
            float x0 = s0[v] + srow[col];
            float x1 = s1[v] + srow[col + 16];
            float mv = fmaxf(x0, x1);
            mv = fmaxf(mv, __shfl_xor(mv, 1, 32));
            mv = fmaxf(mv, __shfl_xor(mv, 2, 32));
            mv = fmaxf(mv, __shfl_xor(mv, 4, 32));
            mv = fmaxf(mv, __shfl_xor(mv, 8, 32));
            const float mnew = fmaxf(mrun[v], mv);
            const float corr = __expf(mrun[v] - mnew);
            mrun[v] = mnew;
            const float p0 = __expf(x0 - mnew);
            const float p1 = __expf(x1 - mnew);
            lrun[v] = lrun[v] * corr + (p0 + p1);   // per-lane partial sum
            acc0[v] *= corr; acc1[v] *= corr; acc2[v] *= corr; acc3[v] *= corr;
            ldsP[wave][v + half][col]      = (_Float16)p0;
            ldsP[wave][v + half][col + 16] = (_Float16)p1;
        }

        // ctx += P(16x32) . V(32x64)
        v16h pa  = load_frag_row(&ldsP[wave][col][0], lane);
        v16h bv0 = load_frag_row(&ldsVT[cur][col][0], lane);
        acc0 = __builtin_amdgcn_wmma_f32_16x16x32_f16(false, pa, false, bv0, (short)0, acc0, false, false);
        bv0 = load_frag_row(&ldsVT[cur][16 + col][0], lane);
        acc1 = __builtin_amdgcn_wmma_f32_16x16x32_f16(false, pa, false, bv0, (short)0, acc1, false, false);
        bv0 = load_frag_row(&ldsVT[cur][32 + col][0], lane);
        acc2 = __builtin_amdgcn_wmma_f32_16x16x32_f16(false, pa, false, bv0, (short)0, acc2, false, false);
        bv0 = load_frag_row(&ldsVT[cur][48 + col][0], lane);
        acc3 = __builtin_amdgcn_wmma_f32_16x16x32_f16(false, pa, false, bv0, (short)0, acc3, false, false);

        // commit next tile's staging
        if (more) {
#if !HAVE_TDM
            *reinterpret_cast<uint4*>(&ldsK[nxt][kk][ch]) = kpre;
#endif
#pragma unroll
            for (int i = 0; i < 8; ++i) ldsVT[nxt][ch + i][kk] = vpre.hlf[i];
#if HAVE_TDM
            if (wave == 0) __builtin_amdgcn_s_wait_tensorcnt(0);
#endif
        }
        __syncthreads();
        cur = nxt;
    }

    // final cross-lane row-sum reduction, normalize, write ctx f16
#pragma unroll
    for (int v = 0; v < 8; ++v) {
        float ls = lrun[v];
        ls += __shfl_xor(ls, 1, 32);
        ls += __shfl_xor(ls, 2, 32);
        ls += __shfl_xor(ls, 4, 32);
        ls += __shfl_xor(ls, 8, 32);
        const float inv = 1.0f / ls;
        const int row = q0 + v + half;
        const size_t base = ((size_t)b * S_LEN + row) * D_MODEL + h * HDIM;
        Cx[base + col]      = (_Float16)(acc0[v] * inv);
        Cx[base + 16 + col] = (_Float16)(acc1[v] * inv);
        Cx[base + 32 + col] = (_Float16)(acc2[v] * inv);
        Cx[base + 48 + col] = (_Float16)(acc3[v] * inv);
    }
}

extern "C" void kernel_launch(void* const* d_in, const int* in_sizes, int n_in,
                              void* d_out, int out_size, void* d_ws, size_t ws_size,
                              hipStream_t stream) {
    const float* q         = (const float*)d_in[0];
    const float* k         = (const float*)d_in[1];
    const float* v         = (const float*)d_in[2];
    const float* structure = (const float*)d_in[3];
    const float* Wq = (const float*)d_in[4];
    const float* bq = (const float*)d_in[5];
    const float* Wk = (const float*)d_in[6];
    const float* bk = (const float*)d_in[7];
    const float* Wv = (const float*)d_in[8];
    const float* bv = (const float*)d_in[9];
    const float* Wo = (const float*)d_in[10];
    const float* bo = (const float*)d_in[11];
    float* out = (float*)d_out;

    const size_t elems = (size_t)BATCH * NHEAD * S_LEN * HDIM;  // 2,097,152
    _Float16* Qf = (_Float16*)d_ws;
    _Float16* Kf = Qf + elems;
    _Float16* Vf = Kf + elems;
    _Float16* Cx = Vf + elems;

    dim3 blk(256);
    dim3 gproj(64, 8);                        // 4096/64 x 512/64
    // 1/sqrt(DK) folded into the Q projection output
    proj_kernel<<<gproj, blk, 0, stream>>>(q, nullptr, Wq, bq, Qf, nullptr, 0.125f, 0);
    proj_kernel<<<gproj, blk, 0, stream>>>(k, nullptr, Wk, bk, Kf, nullptr, 1.0f, 1);
    proj_kernel<<<gproj, blk, 0, stream>>>(v, nullptr, Wv, bv, Vf, nullptr, 1.0f, 2);

    dim3 gattn(S_LEN / 128, BATCH * NHEAD);   // 16 x 16
    attn_kernel<<<gattn, blk, 0, stream>>>(Qf, Kf, Vf, structure, Cx);

    proj_kernel<<<gproj, blk, 0, stream>>>(nullptr, Cx, Wo, bo, nullptr, out, 1.0f, 3);
}
